// GraphReasoningNetwork_20194936225990
// MI455X (gfx1250) — compile-verified, matrix operand order checked by reading, not run on hardware
//
#include <hip/hip_runtime.h>
#include <stdint.h>

#define HDIM 128
#define NODE_IN 256
#define EDGE_IN 64

typedef __bf16 bf16;
typedef __attribute__((ext_vector_type(16))) __bf16 v16bf;
typedef __attribute__((ext_vector_type(8)))  float  v8f;

struct alignas(16) U4 { uint32_t a, b, c, d; };

// ---------------------------------------------------------------------------
// WMMA fragment helpers (CDNA5 wave32, V_WMMA_F32_16X16X32_BF16)
// A-fragment (16x32 bf16): lane L holds row (L&15); its 16 bf16 values are two
// contiguous 8-element chunks at K = k0+sel and K = k0+16+sel, sel = (L&16)?8:0.
// B-fragment uses the transpose-symmetric layout against W^T[n][k] (n-major).
// C/D (16x16 f32): lane L, vgpr v -> row v + ((L&16)?8:0), col (L&15).
// ---------------------------------------------------------------------------
__device__ __forceinline__ v16bf frag_load(const bf16* row, int k0, int sel8) {
  union { v16bf v; U4 q[2]; } u;
  u.q[0] = *(const U4*)(row + k0 + sel8);
  u.q[1] = *(const U4*)(row + k0 + 16 + sel8);
  return u.v;
}

// One 16-row x 128-col layer: all 8 N-tiles kept in registers so each
// A-fragment is loaded from LDS once and reused by 8 consecutive WMMAs.
template <int K>
__device__ __forceinline__ void gemm8(const bf16* __restrict__ A,
                                      const bf16* __restrict__ WT,
                                      const float* __restrict__ bias,
                                      int lane, v8f acc[8]) {
  const int r   = lane & 15;
  const int sel = (lane & 16) ? 8 : 0;
#pragma unroll
  for (int nt = 0; nt < 8; ++nt) {
    float b = bias[nt * 16 + r];
#pragma unroll
    for (int i = 0; i < 8; ++i) acc[nt][i] = b;
  }
  const bf16* arow = A + r * K;
  const bf16* brow = WT + (size_t)r * K;
#pragma unroll 2
  for (int k0 = 0; k0 < K; k0 += 32) {
    v16bf a = frag_load(arow, k0, sel);
#pragma unroll
    for (int nt = 0; nt < 8; ++nt) {
      v16bf b = frag_load(brow + (size_t)nt * 16 * K, k0, sel);
      acc[nt] = __builtin_amdgcn_wmma_f32_16x16x32_bf16(false, a, false, b,
                                                        (short)0, acc[nt],
                                                        false, false);
    }
  }
}

__device__ __forceinline__ void relu_to_hidden(bf16* hid, v8f acc[8], int lane) {
  const int rb = (lane & 16) ? 8 : 0;
#pragma unroll
  for (int nt = 0; nt < 8; ++nt) {
    const int col = nt * 16 + (lane & 15);
#pragma unroll
    for (int v = 0; v < 8; ++v) {
      float x = acc[nt][v] > 0.0f ? acc[nt][v] : 0.0f;
      hid[(rb + v) * HDIM + col] = (bf16)x;
    }
  }
}

// Single 16x16 output tile (used by predictor layer 2 with zero-padded B).
__device__ __forceinline__ v8f gemm_tile(const bf16* A, int K, const bf16* WT,
                                         int ntile, float bias, int lane) {
  v8f c;
#pragma unroll
  for (int i = 0; i < 8; ++i) c[i] = bias;
  const int r   = lane & 15;
  const int sel = (lane & 16) ? 8 : 0;
  const bf16* arow = A + r * K;
  const bf16* brow = WT + (size_t)(ntile * 16 + r) * K;
  for (int k0 = 0; k0 < K; k0 += 32) {
    v16bf a = frag_load(arow, k0, sel);
    v16bf b = frag_load(brow, k0, sel);
    c = __builtin_amdgcn_wmma_f32_16x16x32_bf16(false, a, false, b, (short)0, c,
                                                false, false);
  }
  return c;
}

// ---------------------------------------------------------------------------
// Small utility kernels
// ---------------------------------------------------------------------------
__global__ __launch_bounds__(256) void k_transpose_bf16(const float* __restrict__ w,
                                                        bf16* __restrict__ wt,
                                                        int K, int Ncols) {
  int i = blockIdx.x * 256 + threadIdx.x;
  if (i < K * Ncols) {
    int k = i / Ncols, n = i % Ncols;
    wt[(size_t)n * K + k] = (bf16)w[i];
  }
}

// pr_w2 [128,1] -> padded B^T [16 x 128], row 0 = w2, rows 1..15 = 0
__global__ __launch_bounds__(256) void k_build_pr_w2(const float* __restrict__ w2,
                                                     bf16* __restrict__ wt) {
  int i = blockIdx.x * 256 + threadIdx.x;
  if (i < 16 * HDIM) {
    int n = i / HDIM, k = i % HDIM;
    wt[i] = (n == 0) ? (bf16)w2[k] : (bf16)0.0f;
  }
}

__global__ __launch_bounds__(256) void k_zero_f32(float* __restrict__ p, size_t n) {
  size_t i = (size_t)blockIdx.x * 256 + threadIdx.x;
  if (i < n) p[i] = 0.0f;
}

// ---------------------------------------------------------------------------
// Stage 1: node encoder  [N,256] -> relu -> [N,128]  (bf16 state out)
// ---------------------------------------------------------------------------
__global__ __launch_bounds__(64)
void k_node_encoder(const float* __restrict__ nf,
                    const bf16* __restrict__ w1T, const float* __restrict__ b1,
                    const bf16* __restrict__ w2T, const float* __restrict__ b2,
                    bf16* __restrict__ node_state, int nTiles) {
  __shared__ __align__(16) bf16 A[2][16 * NODE_IN];
  __shared__ __align__(16) bf16 Hid[2][16 * HDIM];
  const int wave = threadIdx.x >> 5, lane = threadIdx.x & 31;
  int tile = blockIdx.x * 2 + wave;
  if (tile >= nTiles) tile = nTiles - 1;
  const int r0 = tile * 16;
  bf16* At = A[wave]; bf16* hid = Hid[wave];

  for (int i = lane; i < 16 * NODE_IN; i += 32) {
    int r = i >> 8, c = i & 255;
    At[i] = (bf16)nf[(size_t)(r0 + r) * NODE_IN + c];
  }
  __syncthreads();
  v8f acc[8];
  gemm8<NODE_IN>(At, w1T, b1, lane, acc);
  relu_to_hidden(hid, acc, lane);
  __syncthreads();
  gemm8<HDIM>(hid, w2T, b2, lane, acc);
  const int rb = (lane & 16) ? 8 : 0;
#pragma unroll
  for (int nt = 0; nt < 8; ++nt) {
    const int col = nt * 16 + (lane & 15);
#pragma unroll
    for (int v = 0; v < 8; ++v)
      node_state[(size_t)(r0 + rb + v) * HDIM + col] = (bf16)acc[nt][v];
  }
}

// ---------------------------------------------------------------------------
// Stage 2: edge encoder  [E,64] -> relu -> [E,128]  (bf16 state out)
// ---------------------------------------------------------------------------
__global__ __launch_bounds__(64)
void k_edge_encoder(const float* __restrict__ ef,
                    const bf16* __restrict__ w1T, const float* __restrict__ b1,
                    const bf16* __restrict__ w2T, const float* __restrict__ b2,
                    bf16* __restrict__ edge_state, int nTiles) {
  __shared__ __align__(16) bf16 A[2][16 * EDGE_IN];
  __shared__ __align__(16) bf16 Hid[2][16 * HDIM];
  const int wave = threadIdx.x >> 5, lane = threadIdx.x & 31;
  int tile = blockIdx.x * 2 + wave;
  if (tile >= nTiles) tile = nTiles - 1;
  const int e0 = tile * 16;
  bf16* At = A[wave]; bf16* hid = Hid[wave];

  for (int i = lane; i < 16 * EDGE_IN; i += 32) {
    int r = i >> 6, c = i & 63;
    At[i] = (bf16)ef[(size_t)(e0 + r) * EDGE_IN + c];
  }
  __syncthreads();
  v8f acc[8];
  gemm8<EDGE_IN>(At, w1T, b1, lane, acc);
  relu_to_hidden(hid, acc, lane);
  __syncthreads();
  gemm8<HDIM>(hid, w2T, b2, lane, acc);
  const int rb = (lane & 16) ? 8 : 0;
#pragma unroll
  for (int nt = 0; nt < 8; ++nt) {
    const int col = nt * 16 + (lane & 15);
#pragma unroll
    for (int v = 0; v < 8; ++v)
      edge_state[(size_t)(e0 + rb + v) * HDIM + col] = (bf16)acc[nt][v];
  }
}

// ---------------------------------------------------------------------------
// Stage 3: message MLP  concat(ns[src], ns[dst], es) [E,384] -> [E,128]
//          + atomic segment-sum into agg[dst]
// ---------------------------------------------------------------------------
__global__ __launch_bounds__(64)
void k_message(const int* __restrict__ ei,
               const bf16* __restrict__ node_state, const bf16* __restrict__ edge_state,
               const bf16* __restrict__ w1T, const float* __restrict__ b1,
               const bf16* __restrict__ w2T, const float* __restrict__ b2,
               float* __restrict__ messages, float* __restrict__ agg,
               int nTiles, int E) {
  __shared__ __align__(16) bf16 A[2][16 * 3 * HDIM];
  __shared__ __align__(16) bf16 Hid[2][16 * HDIM];
  __shared__ int SIdx[2][16], DIdx[2][16];
  const int wave = threadIdx.x >> 5, lane = threadIdx.x & 31;
  int tile = blockIdx.x * 2 + wave;
  if (tile >= nTiles) tile = nTiles - 1;
  const int e0 = tile * 16;
  bf16* At = A[wave]; bf16* hid = Hid[wave];

  if (lane < 16) SIdx[wave][lane] = ei[e0 + lane];
  else           DIdx[wave][lane - 16] = ei[E + e0 + lane - 16];
  __syncthreads();

  // 16 rows x 384 bf16 cols, copied as 16B chunks (48 per row)
  for (int i = lane; i < 16 * 48; i += 32) {
    int r = i / 48, c0 = (i % 48) * 8;
    const bf16* src;
    if (c0 < 128)      src = node_state + (size_t)SIdx[wave][r] * HDIM + c0;
    else if (c0 < 256) src = node_state + (size_t)DIdx[wave][r] * HDIM + (c0 - 128);
    else               src = edge_state + (size_t)(e0 + r) * HDIM + (c0 - 256);
    *(U4*)(At + r * (3 * HDIM) + c0) = *(const U4*)src;
  }
  __syncthreads();
  v8f acc[8];
  gemm8<3 * HDIM>(At, w1T, b1, lane, acc);
  relu_to_hidden(hid, acc, lane);
  __syncthreads();
  gemm8<HDIM>(hid, w2T, b2, lane, acc);
  const int rb = (lane & 16) ? 8 : 0;
#pragma unroll
  for (int nt = 0; nt < 8; ++nt) {
    const int col = nt * 16 + (lane & 15);
#pragma unroll
    for (int v = 0; v < 8; ++v) {
      messages[(size_t)(e0 + rb + v) * HDIM + col] = acc[nt][v];
      atomicAdd(&agg[(size_t)DIdx[wave][rb + v] * HDIM + col], acc[nt][v]);
    }
  }
}

// ---------------------------------------------------------------------------
// Stage 4: node updater  concat(ns, agg) [N,256] -> [N,128] (f32 out + bf16)
// ---------------------------------------------------------------------------
__global__ __launch_bounds__(64)
void k_node_update(const bf16* __restrict__ node_state, const float* __restrict__ agg,
                   const bf16* __restrict__ w1T, const float* __restrict__ b1,
                   const bf16* __restrict__ w2T, const float* __restrict__ b2,
                   float* __restrict__ out_f32, bf16* __restrict__ out_bf,
                   int nTiles) {
  __shared__ __align__(16) bf16 A[2][16 * 2 * HDIM];
  __shared__ __align__(16) bf16 Hid[2][16 * HDIM];
  const int wave = threadIdx.x >> 5, lane = threadIdx.x & 31;
  int tile = blockIdx.x * 2 + wave;
  if (tile >= nTiles) tile = nTiles - 1;
  const int r0 = tile * 16;
  bf16* At = A[wave]; bf16* hid = Hid[wave];

  // cols 0..127: bf16 copy; cols 128..255: f32 agg converted
  for (int i = lane; i < 16 * 16; i += 32) {        // 16B chunks, 16 per row
    int r = i >> 4, c0 = (i & 15) * 8;
    *(U4*)(At + r * (2 * HDIM) + c0) =
        *(const U4*)(node_state + (size_t)(r0 + r) * HDIM + c0);
  }
  for (int i = lane; i < 16 * HDIM; i += 32) {
    int r = i >> 7, c = i & 127;
    At[r * (2 * HDIM) + HDIM + c] = (bf16)agg[(size_t)(r0 + r) * HDIM + c];
  }
  __syncthreads();
  v8f acc[8];
  gemm8<2 * HDIM>(At, w1T, b1, lane, acc);
  relu_to_hidden(hid, acc, lane);
  __syncthreads();
  gemm8<HDIM>(hid, w2T, b2, lane, acc);
  const int rb = (lane & 16) ? 8 : 0;
#pragma unroll
  for (int nt = 0; nt < 8; ++nt) {
    const int col = nt * 16 + (lane & 15);
#pragma unroll
    for (int v = 0; v < 8; ++v) {
      size_t o = (size_t)(r0 + rb + v) * HDIM + col;
      out_f32[o] = acc[nt][v];
      out_bf[o]  = (bf16)acc[nt][v];
    }
  }
}

// ---------------------------------------------------------------------------
// Stage 5: edge updater concat(un[src],un[dst],es,msg) [E,512] -> [E,128]
// ---------------------------------------------------------------------------
__global__ __launch_bounds__(64)
void k_edge_update(const int* __restrict__ ei,
                   const bf16* __restrict__ un_state, const bf16* __restrict__ edge_state,
                   const float* __restrict__ msgs,
                   const bf16* __restrict__ w1T, const float* __restrict__ b1,
                   const bf16* __restrict__ w2T, const float* __restrict__ b2,
                   float* __restrict__ upd_edge, int nTiles, int E) {
  __shared__ __align__(16) bf16 A[2][16 * 4 * HDIM];
  __shared__ __align__(16) bf16 Hid[2][16 * HDIM];
  __shared__ int SIdx[2][16], DIdx[2][16];
  const int wave = threadIdx.x >> 5, lane = threadIdx.x & 31;
  int tile = blockIdx.x * 2 + wave;
  if (tile >= nTiles) tile = nTiles - 1;
  const int e0 = tile * 16;
  bf16* At = A[wave]; bf16* hid = Hid[wave];

  if (lane < 16) SIdx[wave][lane] = ei[e0 + lane];
  else           DIdx[wave][lane - 16] = ei[E + e0 + lane - 16];
  __syncthreads();

  for (int i = lane; i < 16 * 48; i += 32) {        // bf16 cols 0..383
    int r = i / 48, c0 = (i % 48) * 8;
    const bf16* src;
    if (c0 < 128)      src = un_state   + (size_t)SIdx[wave][r] * HDIM + c0;
    else if (c0 < 256) src = un_state   + (size_t)DIdx[wave][r] * HDIM + (c0 - 128);
    else               src = edge_state + (size_t)(e0 + r) * HDIM + (c0 - 256);
    *(U4*)(At + r * (4 * HDIM) + c0) = *(const U4*)src;
  }
  for (int i = lane; i < 16 * HDIM; i += 32) {      // f32 messages -> cols 384..511
    int r = i >> 7, c = i & 127;
    At[r * (4 * HDIM) + 3 * HDIM + c] = (bf16)msgs[(size_t)(e0 + r) * HDIM + c];
  }
  __syncthreads();
  v8f acc[8];
  gemm8<4 * HDIM>(At, w1T, b1, lane, acc);
  relu_to_hidden(hid, acc, lane);
  __syncthreads();
  gemm8<HDIM>(hid, w2T, b2, lane, acc);
  const int rb = (lane & 16) ? 8 : 0;
#pragma unroll
  for (int nt = 0; nt < 8; ++nt) {
    const int col = nt * 16 + (lane & 15);
#pragma unroll
    for (int v = 0; v < 8; ++v)
      upd_edge[(size_t)(e0 + rb + v) * HDIM + col] = acc[nt][v];
  }
}

// ---------------------------------------------------------------------------
// Stage 6: predictor  [E,128] -> relu -> [E,1] (layer-2 via zero-padded B)
// ---------------------------------------------------------------------------
__global__ __launch_bounds__(64)
void k_predict(const float* __restrict__ upd_edge,
               const bf16* __restrict__ w1T, const float* __restrict__ b1,
               const bf16* __restrict__ w2Tpad, const float* __restrict__ b2,
               float* __restrict__ logits, int nTiles) {
  __shared__ __align__(16) bf16 A[2][16 * HDIM];
  __shared__ __align__(16) bf16 Hid[2][16 * HDIM];
  const int wave = threadIdx.x >> 5, lane = threadIdx.x & 31;
  int tile = blockIdx.x * 2 + wave;
  if (tile >= nTiles) tile = nTiles - 1;
  const int e0 = tile * 16;
  bf16* At = A[wave]; bf16* hid = Hid[wave];

  for (int i = lane; i < 16 * HDIM; i += 32) {
    int r = i >> 7, c = i & 127;
    At[i] = (bf16)upd_edge[(size_t)(e0 + r) * HDIM + c];
  }
  __syncthreads();
  v8f acc[8];
  gemm8<HDIM>(At, w1T, b1, lane, acc);
  relu_to_hidden(hid, acc, lane);
  __syncthreads();

  float bias0 = ((lane & 15) == 0) ? b2[0] : 0.0f;
  v8f c = gemm_tile(hid, HDIM, w2Tpad, 0, bias0, lane);
  if ((lane & 15) == 0) {                 // column 0 of D holds the logits
    const int rb = (lane & 16) ? 8 : 0;
#pragma unroll
    for (int v = 0; v < 8; ++v) logits[e0 + rb + v] = c[v];
  }
}

// ---------------------------------------------------------------------------
// Host orchestration
// ---------------------------------------------------------------------------
extern "C" void kernel_launch(void* const* d_in, const int* in_sizes, int n_in,
                              void* d_out, int out_size, void* d_ws, size_t ws_size,
                              hipStream_t stream) {
  const float* node_features = (const float*)d_in[0];
  const float* edge_features = (const float*)d_in[1];
  const int*   edge_index    = (const int*)d_in[2];
  const float* W[24];
  for (int i = 0; i < 24; ++i) W[i] = (const float*)d_in[3 + i];
  // W: [0..3]=ne w1,b1,w2,b2  [4..7]=ee  [8..11]=me  [12..15]=nu  [16..19]=eu  [20..23]=pr

  const int N = in_sizes[0] / NODE_IN;   // 50000
  const int E = in_sizes[1] / EDGE_IN;   // 600000

  float* out      = (float*)d_out;
  float* logits   = out;                                   // [E]
  float* upd_node = logits + E;                            // [N,128]
  float* upd_edge = upd_node + (size_t)N * HDIM;           // [E,128]
  float* messages = upd_edge + (size_t)E * HDIM;           // [E,128]

  char* wsp = (char*)d_ws;
  auto carve = [&](size_t bytes) -> void* {
    void* p = (void*)wsp;
    wsp += (bytes + 255) & ~((size_t)255);
    return p;
  };
  bf16* ne_w1T = (bf16*)carve((size_t)NODE_IN * HDIM * 2);
  bf16* ne_w2T = (bf16*)carve((size_t)HDIM * HDIM * 2);
  bf16* ee_w1T = (bf16*)carve((size_t)EDGE_IN * HDIM * 2);
  bf16* ee_w2T = (bf16*)carve((size_t)HDIM * HDIM * 2);
  bf16* me_w1T = (bf16*)carve((size_t)(3 * HDIM) * HDIM * 2);
  bf16* me_w2T = (bf16*)carve((size_t)HDIM * HDIM * 2);
  bf16* nu_w1T = (bf16*)carve((size_t)(2 * HDIM) * HDIM * 2);
  bf16* nu_w2T = (bf16*)carve((size_t)HDIM * HDIM * 2);
  bf16* eu_w1T = (bf16*)carve((size_t)(4 * HDIM) * HDIM * 2);
  bf16* eu_w2T = (bf16*)carve((size_t)HDIM * HDIM * 2);
  bf16* pr_w1T = (bf16*)carve((size_t)HDIM * HDIM * 2);
  bf16* pr_w2T = (bf16*)carve((size_t)16 * HDIM * 2);
  bf16* node_state_bf = (bf16*)carve((size_t)N * HDIM * 2);
  bf16* edge_state_bf = (bf16*)carve((size_t)E * HDIM * 2);
  bf16* un_bf         = (bf16*)carve((size_t)N * HDIM * 2);
  float* agg          = (float*)carve((size_t)N * HDIM * 4);
  (void)ws_size; (void)n_in; (void)out_size;

  auto tr = [&](const float* w, bf16* wt, int K, int Ncols) {
    int total = K * Ncols;
    k_transpose_bf16<<<(total + 255) / 256, 256, 0, stream>>>(w, wt, K, Ncols);
  };
  tr(W[0],  ne_w1T, NODE_IN, HDIM);
  tr(W[2],  ne_w2T, HDIM, HDIM);
  tr(W[4],  ee_w1T, EDGE_IN, HDIM);
  tr(W[6],  ee_w2T, HDIM, HDIM);
  tr(W[8],  me_w1T, 3 * HDIM, HDIM);
  tr(W[10], me_w2T, HDIM, HDIM);
  tr(W[12], nu_w1T, 2 * HDIM, HDIM);
  tr(W[14], nu_w2T, HDIM, HDIM);
  tr(W[16], eu_w1T, 4 * HDIM, HDIM);
  tr(W[18], eu_w2T, HDIM, HDIM);
  tr(W[20], pr_w1T, HDIM, HDIM);
  k_build_pr_w2<<<(16 * HDIM + 255) / 256, 256, 0, stream>>>(W[22], pr_w2T);

  const int ntN = N / 16, ntE = E / 16;

  k_node_encoder<<<(ntN + 1) / 2, 64, 0, stream>>>(node_features, ne_w1T, W[1],
                                                   ne_w2T, W[3], node_state_bf, ntN);
  k_edge_encoder<<<(ntE + 1) / 2, 64, 0, stream>>>(edge_features, ee_w1T, W[5],
                                                   ee_w2T, W[7], edge_state_bf, ntE);
  size_t aggN = (size_t)N * HDIM;
  k_zero_f32<<<(int)((aggN + 255) / 256), 256, 0, stream>>>(agg, aggN);

  k_message<<<(ntE + 1) / 2, 64, 0, stream>>>(edge_index, node_state_bf, edge_state_bf,
                                              me_w1T, W[9], me_w2T, W[11],
                                              messages, agg, ntE, E);
  k_node_update<<<(ntN + 1) / 2, 64, 0, stream>>>(node_state_bf, agg,
                                                  nu_w1T, W[13], nu_w2T, W[15],
                                                  upd_node, un_bf, ntN);
  k_edge_update<<<(ntE + 1) / 2, 64, 0, stream>>>(edge_index, un_bf, edge_state_bf, messages,
                                                  eu_w1T, W[17], eu_w2T, W[19],
                                                  upd_edge, ntE, E);
  k_predict<<<(ntE + 1) / 2, 64, 0, stream>>>(upd_edge, pr_w1T, W[21],
                                              pr_w2T, W[23], logits, ntE);
}
